// PostProcessor_74543452389400
// MI455X (gfx1250) — compile-verified, matrix operand order checked by reading, not run on hardware
//
#include <hip/hip_runtime.h>
#include <hip/hip_bf16.h>
#include <math.h>

#define NN 2048
#define NC 9
#define FG 8
#define SCORE_THRESH 0.05f
#define NMS_THRESH 0.5f
#define DETS 100
#define CLIPV 4.135166556742356f   // log(1000/16)

typedef __attribute__((ext_vector_type(16))) _Float16 v16h;
typedef __attribute__((ext_vector_type(8)))  float    v8f;

// ---------------------------------------------------------------------------
// K1: softmax over 9 classes + per-class box decode.
// ---------------------------------------------------------------------------
__global__ void decode_kernel(const float* __restrict__ logits,
                              const float* __restrict__ reg,
                              const float* __restrict__ anchors,
                              float* __restrict__ scores,   // [FG][NN]
                              float* __restrict__ boxes)    // [FG][NN][7]
{
    int i = blockIdx.x * blockDim.x + threadIdx.x;
    if (i >= NN) return;
    float lg[NC];
    float m = -INFINITY;
    #pragma unroll
    for (int c = 0; c < NC; ++c) { lg[c] = logits[i * NC + c]; m = fmaxf(m, lg[c]); }
    float sum = 0.f;
    #pragma unroll
    for (int c = 0; c < NC; ++c) { lg[c] = expf(lg[c] - m); sum += lg[c]; }
    float inv = 1.f / sum;

    float xa = anchors[i*7+0], ya = anchors[i*7+1], za = anchors[i*7+2];
    float wa = anchors[i*7+3], la = anchors[i*7+4], ha = anchors[i*7+5];
    float ta = anchors[i*7+6];

    #pragma unroll
    for (int c = 1; c < NC; ++c) {
        const float* r = reg + i * (NC*7) + c * 7;
        int cc = c - 1;
        float cx = r[0] * 0.1f * wa + xa;       // /W_CTR
        float cy = r[1] * 0.1f * la + ya;
        float cz = r[2] * 0.1f * ha + za;
        float w  = expf(fminf(r[3] * 0.2f, CLIPV)) * wa;  // /W_SIZE
        float l  = expf(fminf(r[4] * 0.2f, CLIPV)) * la;
        float h  = expf(fminf(r[5] * 0.2f, CLIPV)) * ha;
        float t  = r[6] * 0.1f + ta;
        float* b = boxes + ((size_t)cc * NN + i) * 7;
        b[0]=cx; b[1]=cy; b[2]=cz; b[3]=w; b[4]=l; b[5]=h; b[6]=t;
        scores[cc * NN + i] = lg[c] * inv;
    }
}

// ---------------------------------------------------------------------------
// K2: rank computation via WMMA. rank[i] = #{j : s[j]>s[i] || (s[j]==s[i] && j<i)}
// = row-sum of the comparison matrix = (cmp-tile 16x32 f16) x (ones 32x16) with
// exact f32 accumulation -> V_WMMA_F32_16X16X32_F16. Scores are pre-masked
// (<=thresh -> -inf) to match jnp.argsort(-where(valid,s,-inf)) stable order.
// 128 blocks x 256 thr (8 waves); wave handles one 16-row tile, 64 WMMAs.
// ---------------------------------------------------------------------------
__global__ void rank_kernel(const float* __restrict__ scores,
                            int* __restrict__ order)      // [FG][NN]: order[rank]=i
{
    __shared__ float s_lds[NN];
    int cls = blockIdx.x >> 4;       // 16 blocks per class
    int grp = blockIdx.x & 15;
    const float* s = scores + cls * NN;
    for (int k = threadIdx.x; k < NN; k += blockDim.x) {
        float v = s[k];
        s_lds[k] = (v > SCORE_THRESH) ? v : -INFINITY;
    }
    __syncthreads();

    int wave = threadIdx.x >> 5;
    int lane = threadIdx.x & 31;
    int i0   = (grp * 8 + wave) * 16;     // tile rows [i0, i0+16)
    int mrow = i0 + (lane & 15);
    int half = lane >> 4;                 // A-layout: lanes16-31 hold K-groups +8
    float si = s_lds[mrow];

    v8f acc = {};
    v16h bones;
    #pragma unroll
    for (int e = 0; e < 16; ++e) bones[e] = (_Float16)1.0f;

    for (int j0 = 0; j0 < NN; j0 += 32) {
        v16h a;
        #pragma unroll
        for (int v = 0; v < 8; ++v) {
            // 16-bit A 16x32 layout: VGPR v<4 -> K = 2v (+8 for hi half);
            // VGPR v>=4 -> K = 16 + 2(v-4) (+8 for hi half).
            int klo = (v < 4 ? 2*v : 16 + 2*(v-4)) + (half ? 8 : 0);
            int j   = j0 + klo;
            float sj0 = s_lds[j], sj1 = s_lds[j + 1];
            a[2*v]   = (_Float16)(((sj0 > si) || (sj0 == si && (j    ) < mrow)) ? 1.0f : 0.0f);
            a[2*v+1] = (_Float16)(((sj1 > si) || (sj1 == si && (j + 1) < mrow)) ? 1.0f : 0.0f);
        }
        acc = __builtin_amdgcn_wmma_f32_16x16x32_f16(
                  false, a, false, bones, (short)0, acc, false, false);
    }
    // D layout: VGPR v, lanes0-15 -> M=v, lanes16-31 -> M=8+v (any N column).
    if (lane == 0) {
        #pragma unroll
        for (int v = 0; v < 8; ++v) {
            int r = (int)(acc[v] + 0.5f);
            order[cls * NN + r] = i0 + v;
        }
    } else if (lane == 16) {
        #pragma unroll
        for (int v = 0; v < 8; ++v) {
            int r = (int)(acc[v] + 0.5f);
            order[cls * NN + r] = i0 + 8 + v;
        }
    }
}

// ---------------------------------------------------------------------------
// K3: gather boxes/scores into sorted order.
// ---------------------------------------------------------------------------
__global__ void gather_kernel(const int* __restrict__ order,
                              const float* __restrict__ scores,
                              const float* __restrict__ boxes,
                              float* __restrict__ sscores,
                              float* __restrict__ sboxes)
{
    int gid = blockIdx.x * blockDim.x + threadIdx.x;   // FG*NN
    if (gid >= FG * NN) return;
    int cls = gid / NN;
    int i   = order[gid];
    sscores[gid] = scores[cls * NN + i];
    const float* b = boxes  + ((size_t)cls * NN + i) * 7;
    float*       o = sboxes + (size_t)gid * 7;
    #pragma unroll
    for (int d = 0; d < 7; ++d) o[d] = b[d];
}

// ---------------------------------------------------------------------------
// K4: greedy NMS, one block per class. AABB corners + volumes in LDS (57 KB of
// the WGP's 320 KB), keep bits = 64 LDS words built with wave32 __ballot.
// Sequential scan: step i suppresses j>i iff IoU>thresh and i alive.
// ---------------------------------------------------------------------------
__global__ void nms_kernel(const float* __restrict__ sscores,
                           const float* __restrict__ sboxes,
                           float* __restrict__ flat)       // [FG*NN] masked scores
{
    __shared__ float X1[NN], X2[NN], Y1[NN], Y2[NN], Z1[NN], Z2[NN], VOL[NN];
    __shared__ unsigned keepw[NN / 32];
    int cls = blockIdx.x;
    const float* bb = sboxes  + (size_t)cls * NN * 7;
    const float* ss = sscores + cls * NN;

    for (int p = threadIdx.x; p < NN; p += blockDim.x) {
        float cx = bb[p*7+0], cy = bb[p*7+1], cz = bb[p*7+2];
        float w  = bb[p*7+3], l  = bb[p*7+4], h  = bb[p*7+5];
        X1[p] = cx - 0.5f * w;  X2[p] = cx + 0.5f * w;
        Y1[p] = cy - 0.5f * l;  Y2[p] = cy + 0.5f * l;
        Z1[p] = cz;             Z2[p] = cz + h;
        VOL[p] = w * l * h;
    }
    for (int base = 0; base < NN; base += blockDim.x) {
        int p = base + threadIdx.x;
        unsigned long long bal = __ballot(ss[p] > SCORE_THRESH);
        if ((threadIdx.x & 31) == 0) keepw[p >> 5] = (unsigned)bal;
    }
    __syncthreads();

    for (int i = 0; i < NN - 1; ++i) {
        if ((keepw[i >> 5] >> (i & 31)) & 1u) {
            float ax1=X1[i], ax2=X2[i], ay1=Y1[i], ay2=Y2[i];
            float az1=Z1[i], az2=Z2[i], va=VOL[i];
            for (int j = i + 1 + threadIdx.x; j < NN; j += blockDim.x) {
                if (!((keepw[j >> 5] >> (j & 31)) & 1u)) continue;
                float ix = fmaxf(fminf(ax2, X2[j]) - fmaxf(ax1, X1[j]), 0.f);
                float iy = fmaxf(fminf(ay2, Y2[j]) - fmaxf(ay1, Y1[j]), 0.f);
                float iz = fmaxf(fminf(az2, Z2[j]) - fmaxf(az1, Z1[j]), 0.f);
                float inter = ix * iy * iz;
                float iou = inter / (va + VOL[j] - inter + 1e-7f);
                if (iou > NMS_THRESH)
                    atomicAnd(&keepw[j >> 5], ~(1u << (j & 31)));
            }
        }
        __syncthreads();
    }
    for (int p = threadIdx.x; p < NN; p += blockDim.x) {
        bool k = (keepw[p >> 5] >> (p & 31)) & 1u;
        flat[cls * NN + p] = k ? ss[p] : -INFINITY;
    }
}

// ---------------------------------------------------------------------------
// K5: global top-100 (value desc, index asc tie-break == lax.top_k), one block.
// Destructively masks the ws flat[] copy; regenerated by K4 every launch.
// Output: boxes[0..700) | scores[700..800) | labels[800..900) | ok[900..1000)
// ---------------------------------------------------------------------------
__global__ void topk_kernel(float* __restrict__ flat,
                            const float* __restrict__ sboxes,
                            float* __restrict__ out)
{
    __shared__ float rv[1024];
    __shared__ int   ri[1024];
    int t = threadIdx.x;
    for (int r = 0; r < DETS; ++r) {
        float bv = -INFINITY; int bi = 0x7fffffff;
        for (int k = t; k < FG * NN; k += 1024) {
            float v = flat[k];
            if (v > bv || (v == bv && k < bi)) { bv = v; bi = k; }
        }
        rv[t] = bv; ri[t] = bi;
        __syncthreads();
        for (int s = 512; s > 0; s >>= 1) {
            if (t < s) {
                float ov = rv[t + s]; int oi = ri[t + s];
                if (ov > rv[t] || (ov == rv[t] && oi < ri[t])) { rv[t] = ov; ri[t] = oi; }
            }
            __syncthreads();
        }
        if (t == 0) {
            float v = rv[0]; int b = ri[0];
            bool fin = (v != -INFINITY);
            out[700 + r] = fin ? v : 0.f;
            out[800 + r] = fin ? (float)(b / NN + 1) : 0.f;   // label = class + 1
            out[900 + r] = fin ? 1.f : 0.f;                   // ok flag
            #pragma unroll
            for (int d = 0; d < 7; ++d)
                out[r * 7 + d] = fin ? sboxes[(size_t)b * 7 + d] : 0.f;
            flat[b] = -INFINITY;
        }
        __syncthreads();
    }
}

// ---------------------------------------------------------------------------
extern "C" void kernel_launch(void* const* d_in, const int* in_sizes, int n_in,
                              void* d_out, int out_size, void* d_ws, size_t ws_size,
                              hipStream_t stream)
{
    const float* logits  = (const float*)d_in[0];  // (2048, 9)
    const float* reg     = (const float*)d_in[1];  // (2048, 63)
    const float* anchors = (const float*)d_in[2];  // (2048, 7)
    float* out = (float*)d_out;
    char*  ws  = (char*)d_ws;

    float* scores  = (float*)(ws);             //  65536 B  [FG][NN]
    float* boxes   = (float*)(ws +   65536);   // 458752 B  [FG][NN][7]
    int*   order   = (int*)  (ws +  524288);   //  65536 B  [FG][NN]
    float* sscores = (float*)(ws +  589824);   //  65536 B
    float* sboxes  = (float*)(ws +  655360);   // 458752 B
    float* flat    = (float*)(ws + 1114112);   //  65536 B   (total ~1.13 MB)

    decode_kernel<<<(NN + 255) / 256, 256, 0, stream>>>(logits, reg, anchors, scores, boxes);
    rank_kernel  <<<FG * 16, 256, 0, stream>>>(scores, order);
    gather_kernel<<<(FG * NN + 255) / 256, 256, 0, stream>>>(order, scores, boxes, sscores, sboxes);
    nms_kernel   <<<FG, 1024, 0, stream>>>(sscores, sboxes, flat);
    topk_kernel  <<<1, 1024, 0, stream>>>(flat, sboxes, out);
}
